// MultiCellLSTM_18176301596875
// MI455X (gfx1250) — matrix operand order repaired and checked by reading, not compiled, not run on hardware
//
#include <hip/hip_runtime.h>
#include <hip/hip_bf16.h>

// ---------------- problem constants ----------------
#define T_STEPS 256
#define BATCH   1024
#define HID     512
#define NBLK    128          // persistent grid: 32 hidden tiles x 4 batch groups
#define NTHR    256          // 8 waves of 32
#define KLD     520          // padded K stride (bf16 elems) in LDS: 1040B -> bank-conflict-free cols

// LDS layout (dynamic shared):
//   wsl  : [4 cells][4 gates][16 cols][KLD] bf16   = 266240 B
//   biasS: [4 cells][64]            f32            =   1024 B
//   wihS : [4 cells][64][8]         f32            =   8192 B
#define WBYTES   (4*4*16*KLD*2)
#define SMEM_SZ  (WBYTES + 4*64*4 + 4*64*8*4)

// workspace layout
#define H0_OFF   0
#define H1_OFF   (BATCH*HID*2)            // 1 MB each (bf16)
#define C_OFF    (BATCH*HID*4)            // after both h bufs
#define BAR_OFF  (C_OFF + BATCH*HID*4)

typedef __attribute__((ext_vector_type(16))) __bf16 v16bf;
typedef __attribute__((ext_vector_type(8)))  float  v8f;

struct CellParams {
    const float* wih[4];
    const float* whh[4];
    const float* bih[4];
    const float* bhh[4];
};

__device__ __forceinline__ float fsigm(float x) {
    return 1.0f / (1.0f + __expf(-x));           // v_exp_f32 (trans, co-executes with WMMA)
}
__device__ __forceinline__ float ftanh(float x) {
    float e = __expf(2.0f * x);
    return 1.0f - 2.0f / (e + 1.0f);             // safe at +/-inf
}

__device__ __forceinline__ v16bf load_a_frag(const __bf16* __restrict__ h,
                                             int row, int kBase, int lane) {
    // A 16x32 bf16 layout: lanes 0-15 -> M=lane, K = kBase+{0..7,16..23}
    //                      lanes16-31 -> M=lane-16, K = kBase+{8..15,24..31}
    const __bf16* p = h + (size_t)row * HID + kBase + ((lane >= 16) ? 8 : 0);
    union { uint4 q[2]; v16bf v; } u;
    u.q[0] = *(const uint4*)(p);        // 8 bf16
    u.q[1] = *(const uint4*)(p + 16);   // 8 bf16, K+16
    return u.v;
}

__device__ __forceinline__ v16bf load_b_frag(const __bf16* __restrict__ wslice,
                                             int kBase, int lane) {
    // B 32x16 bf16: col = lane%16, lanes0-15 K=kBase+{0..7,16..23}, lanes16-31 K=kBase+{8..15,24..31}
    const __bf16* p = wslice + (size_t)(lane & 15) * KLD + kBase + ((lane >= 16) ? 8 : 0);
    union { uint4 q[2]; v16bf v; } u;
    u.q[0] = *(const uint4*)(p);        // ds_load_b128
    u.q[1] = *(const uint4*)(p + 16);
    return u.v;
}

__device__ __forceinline__ void grid_barrier(int* cnt, int* gen) {
    __syncthreads();
    if (threadIdx.x == 0) {
        __threadfence();   // make h/c stores device-visible
        int g = __hip_atomic_load(gen, __ATOMIC_RELAXED, __HIP_MEMORY_SCOPE_AGENT);
        int a = __hip_atomic_fetch_add(cnt, 1, __ATOMIC_ACQ_REL, __HIP_MEMORY_SCOPE_AGENT);
        if (a == NBLK - 1) {
            __hip_atomic_store(cnt, 0, __ATOMIC_RELAXED, __HIP_MEMORY_SCOPE_AGENT);
            __hip_atomic_fetch_add(gen, 1, __ATOMIC_RELEASE, __HIP_MEMORY_SCOPE_AGENT);
        } else {
            while (__hip_atomic_load(gen, __ATOMIC_ACQUIRE, __HIP_MEMORY_SCOPE_AGENT) == g)
                __builtin_amdgcn_s_sleep(2);
        }
    }
    __syncthreads();
}

__global__ void lstm_init(__bf16* h0, __bf16* h1, float* c, int* bar) {
    int i = blockIdx.x * blockDim.x + threadIdx.x;
    if (i < BATCH * HID) {
        h0[i] = (__bf16)0.0f;
        h1[i] = (__bf16)0.0f;
        c[i]  = 0.0f;
    }
    if (i < 2) bar[i] = 0;
}

__global__ void __launch_bounds__(NTHR)
lstm_persistent(const float* __restrict__ x_bvp,
                const float* __restrict__ x_acc,
                const float* __restrict__ x_eda,
                const float* __restrict__ x_hr,
                CellParams P,
                __bf16* __restrict__ hbuf0,
                __bf16* __restrict__ hbuf1,
                float*  __restrict__ cbuf,
                int*    __restrict__ bar,
                float*  __restrict__ out) {
    extern __shared__ char smem_raw[];
    __bf16* wsl   = (__bf16*)smem_raw;                       // [4][4][16][KLD]
    float*  biasS = (float*)(smem_raw + WBYTES);             // [4][64]
    float*  wihS  = (float*)(smem_raw + WBYTES + 4*64*4);    // [4][64][8]

    const int hidTile  = blockIdx.x >> 2;      // 0..31
    const int bGroup   = blockIdx.x & 3;       // 0..3
    const int wave     = threadIdx.x >> 5;     // 0..7
    const int lane     = threadIdx.x & 31;
    const int colBase  = hidTile * 16;
    const int colLocal = lane & 15;
    const int halfSel  = lane >> 4;
    const int btile0   = bGroup * 16 + wave * 2;   // two 16-row batch tiles per wave
    const int rowBase0 = btile0 * 16;
    const int rowBase1 = rowBase0 + 16;

    // ---- one-time: stage this block's weight slices for all 4 cells into LDS (f32 -> bf16)
    for (int i = threadIdx.x; i < 4 * 4 * 16 * 512; i += NTHR) {
        int cell = i >> 15;           // / 32768
        int rem  = i & 32767;
        int g    = rem >> 13;         // / 8192
        int rem2 = rem & 8191;
        int c    = rem2 >> 9;         // / 512
        int k    = rem2 & 511;
        int row  = g * HID + colBase + c;            // gate order i,f,g,o along 4H
        float w  = P.whh[cell][(size_t)row * HID + k];
        wsl[((cell * 4 + g) * 16 + c) * KLD + k] = (__bf16)w;
    }
    for (int i = threadIdx.x; i < 4 * 64; i += NTHR) {
        int cell = i >> 6, cc = i & 63;
        int g = cc >> 4, c = cc & 15;
        int row = g * HID + colBase + c;
        biasS[i] = P.bih[cell][row] + P.bhh[cell][row];
        int nInC = (cell == 0) ? 1 : (cell == 1) ? 4 : (cell == 2) ? 5 : 6;
        for (int k = 0; k < 8; ++k)
            wihS[i * 8 + k] = (k < nInC) ? P.wih[cell][row * nInC + k] : 0.0f;
    }
    __syncthreads();

    const int colG = colBase + colLocal;

    for (int t = 0; t < T_STEPS; ++t) {
        // cell-selection schedule (period 64)
        int cell, nIn;
        if      ((t & 63) == 0) { cell = 3; nIn = 6; }
        else if ((t & 15) == 0) { cell = 2; nIn = 5; }
        else if ((t &  1) == 0) { cell = 1; nIn = 4; }
        else                    { cell = 0; nIn = 1; }

        const __bf16* h_in  = (t & 1) ? hbuf1 : hbuf0;
        __bf16*       h_out = (t & 1) ? hbuf0 : hbuf1;
        const __bf16* wcell = wsl + (size_t)cell * (4 * 16 * KLD);

        // ---- recurrent GEMM: 4 gates x 2 batch tiles, K = 512 in 16 chunks of 32
        v8f acc[4][2] = {};
        #pragma unroll 4
        for (int kc = 0; kc < 16; ++kc) {
            const int kBase = kc * 32;
            v16bf a0 = load_a_frag(h_in, rowBase0 + colLocal, kBase, lane);
            v16bf a1 = load_a_frag(h_in, rowBase1 + colLocal, kBase, lane);
            #pragma unroll
            for (int g = 0; g < 4; ++g) {
                v16bf b = load_b_frag(wcell + (size_t)g * (16 * KLD), kBase, lane);
                acc[g][0] = __builtin_amdgcn_wmma_f32_16x16x32_bf16(
                    false, a0, false, b, (short)0, acc[g][0], false, false);
                acc[g][1] = __builtin_amdgcn_wmma_f32_16x16x32_bf16(
                    false, a1, false, b, (short)0, acc[g][1], false, false);
            }
        }

        // ---- fused epilogue: bias + x@Wih^T + activations + state update
        const float bI = biasS[cell * 64 +  0 + colLocal];
        const float bF = biasS[cell * 64 + 16 + colLocal];
        const float bG = biasS[cell * 64 + 32 + colLocal];
        const float bO = biasS[cell * 64 + 48 + colLocal];

        #pragma unroll
        for (int ti = 0; ti < 2; ++ti) {
            #pragma unroll
            for (int r = 0; r < 8; ++r) {
                const int mLocal = r + 8 * halfSel;
                const int bRow   = (btile0 + ti) * 16 + mLocal;

                // assemble x[t] for this batch row (broadcast across 16 lanes)
                float xv[6] = {0.f, 0.f, 0.f, 0.f, 0.f, 0.f};
                xv[0] = x_bvp[(size_t)t * BATCH + bRow];
                if (cell >= 1) {
                    size_t ab = ((size_t)(t >> 1) * BATCH + bRow) * 3;
                    xv[1] = x_acc[ab + 0]; xv[2] = x_acc[ab + 1]; xv[3] = x_acc[ab + 2];
                }
                if (cell >= 2) xv[4] = x_eda[(size_t)(t >> 4) * BATCH + bRow];
                if (cell == 3) xv[5] = x_hr [(size_t)(t >> 6) * BATCH + bRow];

                float xd[4];
                #pragma unroll
                for (int g = 0; g < 4; ++g) {
                    const float* wr = wihS + ((cell * 4 + g) * 16 + colLocal) * 8;
                    float s = 0.0f;
                    for (int k = 0; k < nIn; ++k) s += xv[k] * wr[k];
                    xd[g] = s;
                }

                float gi = acc[0][ti][r] + bI + xd[0];
                float gf = acc[1][ti][r] + bF + xd[1];
                float gg = acc[2][ti][r] + bG + xd[2];
                float go = acc[3][ti][r] + bO + xd[3];

                const size_t idx = (size_t)bRow * HID + colG;
                float c_new = fsigm(gf) * cbuf[idx] + fsigm(gi) * ftanh(gg);
                float h_new = fsigm(go) * ftanh(c_new);

                cbuf[idx]  = c_new;
                h_out[idx] = (__bf16)h_new;
                out[(size_t)t * (BATCH * HID) + idx] = h_new;
            }
        }

        if (t != T_STEPS - 1) grid_barrier(bar, bar + 1);
    }
}

extern "C" void kernel_launch(void* const* d_in, const int* in_sizes, int n_in,
                              void* d_out, int out_size, void* d_ws, size_t ws_size,
                              hipStream_t stream) {
    const float* x_bvp = (const float*)d_in[0];
    const float* x_acc = (const float*)d_in[1];
    const float* x_eda = (const float*)d_in[2];
    const float* x_hr  = (const float*)d_in[3];

    CellParams P;
    // params dict order '1','4','5','6', each (W_ih, W_hh, b_ih, b_hh)
    for (int c = 0; c < 4; ++c) {
        P.wih[c] = (const float*)d_in[4 + 4 * c + 0];
        P.whh[c] = (const float*)d_in[4 + 4 * c + 1];
        P.bih[c] = (const float*)d_in[4 + 4 * c + 2];
        P.bhh[c] = (const float*)d_in[4 + 4 * c + 3];
    }

    char* ws = (char*)d_ws;
    __bf16* h0  = (__bf16*)(ws + H0_OFF);
    __bf16* h1  = (__bf16*)(ws + H1_OFF);
    float*  c   = (float*)(ws + C_OFF);
    int*    bar = (int*)(ws + BAR_OFF);
    float*  out = (float*)d_out;

    (void)hipFuncSetAttribute((const void*)lstm_persistent,
                              hipFuncAttributeMaxDynamicSharedMemorySize, SMEM_SZ);

    lstm_init<<<(BATCH * HID + NTHR - 1) / NTHR, NTHR, 0, stream>>>(h0, h1, c, bar);
    lstm_persistent<<<NBLK, NTHR, SMEM_SZ, stream>>>(
        x_bvp, x_acc, x_eda, x_hr, P, h0, h1, c, bar, out);
}
